// ImageDetector_44641890075157
// MI455X (gfx1250) — compile-verified, matrix operand order checked by reading, not run on hardware
//
#include <hip/hip_runtime.h>
#include <hip/hip_bf16.h>

typedef __attribute__((ext_vector_type(16))) __bf16 v16bf;
typedef __attribute__((ext_vector_type(8)))  float  v8f;

#define BN_EPS 1e-5f
#define NSPLIT 16

__device__ __forceinline__ unsigned short f32_to_bf16_rne(float f) {
    unsigned int u = __float_as_uint(f);
    return (unsigned short)((u + 0x7FFFu + ((u >> 16) & 1u)) >> 16);
}

// ---------------------------------------------------------------------------
// Weight packer: fp32 (Cout, Cin*K*K) -> bf16 A-fragments
// layout: [mtile][ktile][lane 0..31][16 bf16]  (per-lane contiguous 32B)
// A 16-bit 16x32 layout (ISA 7.12.2): lane = M%16 (+16 for K-half),
// VGPR v<4 holds K = half*8 + 2v + p, v>=4 holds K = 16 + half*8 + 2(v-4)+p
// ---------------------------------------------------------------------------
__global__ void pack_w_kernel(const float* __restrict__ w, unsigned short* __restrict__ wp,
                              int Cout, int Ck, int Ktiles, long total) {
    long idx = (long)blockIdx.x * blockDim.x + threadIdx.x;
    if (idx >= total) return;
    int e    = (int)(idx & 15);
    int lane = (int)((idx >> 4) & 31);
    long t   = idx >> 9;
    int kt   = (int)(t % Ktiles);
    int mt   = (int)(t / Ktiles);
    int m    = mt * 16 + (lane & 15);
    int half = lane >> 4;
    int vv = e >> 1, p = e & 1;
    int kl = (vv < 4) ? (half * 8 + vv * 2 + p) : (16 + half * 8 + (vv - 4) * 2 + p);
    int k  = kt * 32 + kl;
    float v = 0.f;
    if (m < Cout && k < Ck) v = w[(long)m * Ck + k];
    wp[idx] = f32_to_bf16_rne(v);
}

// ---------------------------------------------------------------------------
// Generic implicit-GEMM conv (any k/stride). One wave: 64(Cout) x 16(pixel),
// 4x v_wmma_f32_16x16x32_bf16 per K-step. Pixel decomposition hoisted out of
// the K-loop (no div/mod in the hot loop).
// ---------------------------------------------------------------------------
__global__ __launch_bounds__(256) void conv_wmma_kernel(
    const float* __restrict__ in, const unsigned short* __restrict__ wp,
    float* __restrict__ out,
    int N, int Cin, int Hin, int Win,
    int Cout, int Hout, int Wout,
    int ks, int stride, int pad, int Ktiles)
{
    __shared__ unsigned short ldsB[8][32][16];
    const int lane = threadIdx.x & 31;
    const int wave = threadIdx.x >> 5;
    const long totalPix = (long)N * Hout * Wout;
    const long ptile = (long)blockIdx.x * 8 + wave;
    const int  mt0   = blockIdx.y * 4;        // first of 4 M-tiles (pack is padded)
    const int  KK = ks * ks;
    const int  Ck = Cin * KK;

    // hoisted per-pixel decomposition (invariant across K-steps)
    int iys[16], ixs[16], nci[16];
    #pragma unroll
    for (int n = 0; n < 16; ++n) {
        long pix = ptile * 16 + n;
        if (pix >= totalPix) pix = totalPix - 1;   // clamp; columns masked at store
        int ox = (int)(pix % Wout);
        long tt = pix / Wout;
        int oy = (int)(tt % Hout);
        int nb = (int)(tt / Hout);
        iys[n] = oy * stride - pad;
        ixs[n] = ox * stride - pad;
        nci[n] = nb * Cin;
    }

    v8f acc0 = {0.f,0.f,0.f,0.f,0.f,0.f,0.f,0.f};
    v8f acc1 = acc0, acc2 = acc0, acc3 = acc0;

    for (int kt = 0; kt < Ktiles; ++kt) {
        // ---- stage B tile: lane handles k-row (kt*32+lane) for 16 pixels ----
        int k = kt * 32 + lane;
        bool kvalid = (k < Ck);
        int ci = 0, ky = 0, kx = 0;
        if (kvalid) { ci = k / KK; int r = k - ci * KK; ky = r / ks; kx = r - ky * ks; }
        #pragma unroll
        for (int n = 0; n < 16; ++n) {
            float v = 0.f;
            int iy = iys[n] + ky;
            int ix = ixs[n] + kx;
            if (kvalid && (unsigned)iy < (unsigned)Hin && (unsigned)ix < (unsigned)Win)
                v = in[((long)(nci[n] + ci) * Hin + iy) * Win + ix];
            ldsB[wave][lane][n] = f32_to_bf16_rne(v);
        }
        __syncthreads();

        // ---- load 4 A fragments (8x global_load_b128) ----
        union Frag { v16bf v; unsigned short s[16]; uint4 q[2]; };
        Frag a0, a1, a2, a3, b;
        {
            const long stepK = (long)Ktiles * 512;   // elements per mtile
            const unsigned short* apb = wp + (((long)mt0 * Ktiles + kt) * 32 + lane) * 16;
            const uint4* p0 = (const uint4*)(apb);
            const uint4* p1 = (const uint4*)(apb + stepK);
            const uint4* p2 = (const uint4*)(apb + 2 * stepK);
            const uint4* p3 = (const uint4*)(apb + 3 * stepK);
            a0.q[0] = p0[0]; a0.q[1] = p0[1];
            a1.q[0] = p1[0]; a1.q[1] = p1[1];
            a2.q[0] = p2[0]; a2.q[1] = p2[1];
            a3.q[0] = p3[0]; a3.q[1] = p3[1];
            if (kt + 1 < Ktiles)   // gfx1250 global_prefetch for next K-step's A data
                __builtin_prefetch((const void*)(apb + 512), 0, 0);
        }
        const int nn = lane & 15, half = lane >> 4;
        #pragma unroll
        for (int e = 0; e < 16; ++e) {
            int vv = e >> 1, p = e & 1;
            int kl = (vv < 4) ? (half * 8 + vv * 2 + p) : (16 + half * 8 + (vv - 4) * 2 + p);
            b.s[e] = ldsB[wave][kl][nn];
        }
        __syncthreads();

        acc0 = __builtin_amdgcn_wmma_f32_16x16x32_bf16(false, a0.v, false, b.v, (short)0, acc0, false, false);
        acc1 = __builtin_amdgcn_wmma_f32_16x16x32_bf16(false, a1.v, false, b.v, (short)0, acc1, false, false);
        acc2 = __builtin_amdgcn_wmma_f32_16x16x32_bf16(false, a2.v, false, b.v, (short)0, acc2, false, false);
        acc3 = __builtin_amdgcn_wmma_f32_16x16x32_bf16(false, a3.v, false, b.v, (short)0, acc3, false, false);
    }

    // ---- epilogue: C 16x16 f32 layout -> NCHW fp32 ----
    const int nn = lane & 15, half = lane >> 4;
    long pix = ptile * 16 + nn;
    if (pix < totalPix) {
        int ox = (int)(pix % Wout);
        long tt = pix / Wout;
        int oy = (int)(tt % Hout);
        int nb = (int)(tt / Hout);
        long chw = (long)Hout * Wout;
        long obase = (long)nb * Cout * chw + (long)oy * Wout + ox;
        #pragma unroll
        for (int j = 0; j < 4; ++j) {
            v8f acc = (j == 0) ? acc0 : (j == 1) ? acc1 : (j == 2) ? acc2 : acc3;
            #pragma unroll
            for (int r = 0; r < 8; ++r) {
                int co = (mt0 + j) * 16 + half * 8 + r;
                if (co < Cout)
                    out[obase + (long)co * chw] = acc[r];
            }
        }
    }
}

// ---------------------------------------------------------------------------
// 1x1 / stride-1 conv (Cin % 32 == 0): pure GEMM. Interior tiles stage the
// 32x16 fp32 B tile with four global_load_async_to_lds_b128 per lane
// (ASYNCcnt-tracked CDNA5 async copy); edge tiles use a scalar fallback.
// The branch is wave-uniform, LDS regions are per-wave (no block barrier).
// ---------------------------------------------------------------------------
__global__ __launch_bounds__(256) void conv1x1_wmma_kernel(
    const float* __restrict__ in, const unsigned short* __restrict__ wp,
    float* __restrict__ out,
    int N, int Cin, long HW, int Cout, int Ktiles)
{
    __shared__ float ldsF[8][32][16];          // this kernel's only LDS object -> offset 0
    const int lane = threadIdx.x & 31;
    const int wave = threadIdx.x >> 5;
    const long totalPix = (long)N * HW;
    const long p0 = ((long)blockIdx.x * 8 + wave) * 16;
    const int  mt0 = blockIdx.y * 4;

    long hw0 = p0 % HW;
    int  nb0 = (int)(p0 / HW);
    const bool fast = (p0 + 16 <= totalPix) && (hw0 + 16 <= HW);
    const float* ibase = in + (long)nb0 * Cin * HW + hw0;   // (nb0, ci=0, hw0)
    const unsigned ldsbase = (unsigned)wave * (32u * 16u * 4u);

    v8f acc0 = {0.f,0.f,0.f,0.f,0.f,0.f,0.f,0.f};
    v8f acc1 = acc0, acc2 = acc0, acc3 = acc0;

    for (int kt = 0; kt < Ktiles; ++kt) {
        const int c0 = kt * 32;
        if (fast) {
            // WAR guard: prior LDS reads must drain before async overwrites
            asm volatile("s_wait_dscnt 0x0" ::: "memory");
            #pragma unroll
            for (int j = 0; j < 4; ++j) {
                int L = j * 32 + lane;                 // 16-byte chunk index in tile
                int row = L >> 2, col4 = (L & 3) << 2;
                unsigned lds = ldsbase + (unsigned)L * 16u;
                unsigned long long ga =
                    (unsigned long long)(uintptr_t)(ibase + (long)(c0 + row) * HW + col4);
                asm volatile("global_load_async_to_lds_b128 %0, %1, off"
                             :: "v"(lds), "v"(ga) : "memory");
            }
            asm volatile("s_wait_asynccnt 0x0" ::: "memory");
        } else {
            for (int n = 0; n < 16; ++n) {
                long pix = p0 + n;
                float v = 0.f;
                if (pix < totalPix) {
                    long hw = pix % HW;
                    int nb = (int)(pix / HW);
                    v = in[((long)nb * Cin + (c0 + lane)) * HW + hw];
                }
                ldsF[wave][lane][n] = v;
            }
        }

        union Frag { v16bf v; unsigned short s[16]; uint4 q[2]; };
        Frag a0, a1, a2, a3, b;
        {
            const long stepK = (long)Ktiles * 512;
            const unsigned short* apb = wp + (((long)mt0 * Ktiles + kt) * 32 + lane) * 16;
            const uint4* q0 = (const uint4*)(apb);
            const uint4* q1 = (const uint4*)(apb + stepK);
            const uint4* q2 = (const uint4*)(apb + 2 * stepK);
            const uint4* q3 = (const uint4*)(apb + 3 * stepK);
            a0.q[0] = q0[0]; a0.q[1] = q0[1];
            a1.q[0] = q1[0]; a1.q[1] = q1[1];
            a2.q[0] = q2[0]; a2.q[1] = q2[1];
            a3.q[0] = q3[0]; a3.q[1] = q3[1];
            if (kt + 1 < Ktiles)
                __builtin_prefetch((const void*)(apb + 512), 0, 0);
        }
        const int nn = lane & 15, half = lane >> 4;
        #pragma unroll
        for (int e = 0; e < 16; ++e) {
            int vv = e >> 1, p = e & 1;
            int kl = (vv < 4) ? (half * 8 + vv * 2 + p) : (16 + half * 8 + (vv - 4) * 2 + p);
            b.s[e] = f32_to_bf16_rne(ldsF[wave][kl][nn]);
        }

        acc0 = __builtin_amdgcn_wmma_f32_16x16x32_bf16(false, a0.v, false, b.v, (short)0, acc0, false, false);
        acc1 = __builtin_amdgcn_wmma_f32_16x16x32_bf16(false, a1.v, false, b.v, (short)0, acc1, false, false);
        acc2 = __builtin_amdgcn_wmma_f32_16x16x32_bf16(false, a2.v, false, b.v, (short)0, acc2, false, false);
        acc3 = __builtin_amdgcn_wmma_f32_16x16x32_bf16(false, a3.v, false, b.v, (short)0, acc3, false, false);
    }

    const int nn = lane & 15, half = lane >> 4;
    long pix = p0 + nn;
    if (pix < totalPix) {
        long hw = pix % HW;
        int nb = (int)(pix / HW);
        long obase = (long)nb * Cout * HW + hw;
        #pragma unroll
        for (int j = 0; j < 4; ++j) {
            v8f acc = (j == 0) ? acc0 : (j == 1) ? acc1 : (j == 2) ? acc2 : acc3;
            #pragma unroll
            for (int r = 0; r < 8; ++r) {
                int co = (mt0 + j) * 16 + half * 8 + r;
                if (co < Cout)
                    out[obase + (long)co * HW] = acc[r];
            }
        }
    }
}

// ---------------------------------------------------------------------------
// BatchNorm (training mode): per-channel mean/var over N*H*W, split reduction
// (deterministic: fixed split, no float atomics)
// ---------------------------------------------------------------------------
__global__ __launch_bounds__(256) void bn_stats_part_kernel(const float* __restrict__ x,
    float* __restrict__ part, long HW, int N, long CHW) {
    int c = blockIdx.x, sp = blockIdx.y;
    long cnt = (long)N * HW;
    long lo = cnt * sp / NSPLIT, hi = cnt * (sp + 1) / NSPLIT;
    float s = 0.f, s2 = 0.f;
    for (long i = lo + threadIdx.x; i < hi; i += 256) {
        long n = i / HW, r = i - n * HW;
        float v = x[n * CHW + (long)c * HW + r];
        s += v; s2 += v * v;
    }
    __shared__ float sh0[256], sh1[256];
    sh0[threadIdx.x] = s; sh1[threadIdx.x] = s2;
    __syncthreads();
    for (int o = 128; o > 0; o >>= 1) {
        if ((int)threadIdx.x < o) {
            sh0[threadIdx.x] += sh0[threadIdx.x + o];
            sh1[threadIdx.x] += sh1[threadIdx.x + o];
        }
        __syncthreads();
    }
    if (threadIdx.x == 0) {
        part[((long)c * NSPLIT + sp) * 2]     = sh0[0];
        part[((long)c * NSPLIT + sp) * 2 + 1] = sh1[0];
    }
}

__global__ void bn_stats_fin_kernel(const float* __restrict__ part, float* __restrict__ stats,
                                    long cnt, int C) {
    int c = blockIdx.x * blockDim.x + threadIdx.x;
    if (c >= C) return;
    float s = 0.f, s2 = 0.f;
    for (int sp = 0; sp < NSPLIT; ++sp) {
        s  += part[((long)c * NSPLIT + sp) * 2];
        s2 += part[((long)c * NSPLIT + sp) * 2 + 1];
    }
    float inv = 1.f / (float)cnt;
    float mean = s * inv;
    stats[2 * c]     = mean;
    stats[2 * c + 1] = s2 * inv - mean * mean;
}

// BN apply + ReLU, optional fused residual add (after ReLU, matching _res)
__global__ void bn_apply_kernel(const float* __restrict__ raw, const float* __restrict__ stats,
    const float* __restrict__ g, const float* __restrict__ b, const float* __restrict__ addsrc,
    float* __restrict__ out, int C, long HW, long total) {
    long i = (long)blockIdx.x * blockDim.x + threadIdx.x;
    if (i >= total) return;
    int c = (int)((i / HW) % C);
    float mean = stats[2 * c], var = stats[2 * c + 1];
    float v = (raw[i] - mean) * rsqrtf(var + BN_EPS) * g[c] + b[c];
    v = fmaxf(v, 0.f);
    if (addsrc) v += addsrc[i];
    out[i] = v;
}

// dst(B,C,2H,2W) += nearest-up2( src(B,C,H,W) )
__global__ void up2_add_kernel(float* __restrict__ dst, const float* __restrict__ src,
                               int Hs, int Ws, long total) {
    long i = (long)blockIdx.x * blockDim.x + threadIdx.x;
    if (i >= total) return;
    int Wd = Ws * 2, Hd = Hs * 2;
    int x = (int)(i % Wd); long t = i / Wd;
    int y = (int)(t % Hd); t /= Hd;              // t = b*C + c
    dst[i] += src[(t * Hs + (y >> 1)) * Ws + (x >> 1)];
}

// head: (B,255,H,W) + bias -> (B,3,H,W,85)
__global__ void head_out_kernel(const float* __restrict__ raw, const float* __restrict__ pb,
                                float* __restrict__ out, int H, int W, long total) {
    long i = (long)blockIdx.x * blockDim.x + threadIdx.x;
    if (i >= total) return;
    int x = (int)(i % W); long t = i / W;
    int y = (int)(t % H); t /= H;
    int ch = (int)(t % 255); int b = (int)(t / 255);
    int a = ch / 85, c = ch - a * 85;
    out[((((long)b * 3 + a) * H + y) * W + x) * 85 + c] = raw[i] + pb[ch];
}

// ===========================================================================
// Host-side orchestration
// ===========================================================================
static inline int cdiv_i(int a, int b) { return (a + b - 1) / b; }

struct Arena {
    char* base; size_t cap; size_t off;
    void reset() { off = 0; }
    void* alloc_bytes(size_t bytes) {
        bytes = (bytes + 255) & ~(size_t)255;
        if (off + bytes > cap) off = 0;   // wrap guard: stay in-bounds regardless of ws_size
        void* p = base + off; off += bytes; return p;
    }
    float* allocf(size_t n) { return (float*)alloc_bytes(n * sizeof(float)); }
    unsigned short* alloch(size_t n) { return (unsigned short*)alloc_bytes(n * sizeof(unsigned short)); }
};

struct Tensor { float* p; int N, C, H, W; long numel() const { return (long)N * C * H * W; } };
struct CbrP  { const float* bb; const float* g; const float* w; int cin, cout, k; };
struct ResP  { CbrP a, b; };
struct HeadP { CbrP c0, c1; const float* pb; const float* pw; };

static Tensor run_conv(hipStream_t st, Arena& wsA, Arena& outA, const Tensor& in,
                       const float* w, int cout, int k, int stride) {
    int pad = k / 2;
    int Hout = (in.H + 2 * pad - k) / stride + 1;
    int Wout = (in.W + 2 * pad - k) / stride + 1;
    int Ck = in.C * k * k;
    int Ktiles = cdiv_i(Ck, 32);
    int Mquads = cdiv_i(cout, 64);               // wave covers 4 M-tiles
    int MtilesPad = Mquads * 4;
    long wtot = (long)MtilesPad * Ktiles * 512;
    unsigned short* wp = wsA.alloch((size_t)wtot);
    pack_w_kernel<<<(unsigned)((wtot + 255) / 256), 256, 0, st>>>(w, wp, cout, Ck, Ktiles, wtot);

    Tensor out{outA.allocf((size_t)in.N * cout * Hout * Wout), in.N, cout, Hout, Wout};
    long totalPix = (long)in.N * Hout * Wout;
    int Ptiles = (int)((totalPix + 15) / 16);
    dim3 grid((unsigned)cdiv_i(Ptiles, 8), (unsigned)Mquads, 1);
    if (k == 1 && stride == 1 && (in.C % 32) == 0) {
        conv1x1_wmma_kernel<<<grid, 256, 0, st>>>(in.p, wp, out.p, in.N, in.C,
                                                  (long)Hout * Wout, cout, Ktiles);
    } else {
        conv_wmma_kernel<<<grid, 256, 0, st>>>(in.p, wp, out.p, in.N, in.C, in.H, in.W,
                                               cout, Hout, Wout, k, stride, pad, Ktiles);
    }
    return out;
}

static Tensor cbr(hipStream_t st, Arena& S, Arena& outA, float* stats, const Tensor& in,
                  const CbrP& p, int stride, const float* addsrc = nullptr) {
    Tensor raw = run_conv(st, S, outA, in, p.w, p.cout, p.k, stride);
    long HW = (long)raw.H * raw.W;
    long cnt = (long)raw.N * HW;
    float* part = S.allocf((size_t)raw.C * NSPLIT * 2);
    bn_stats_part_kernel<<<dim3((unsigned)raw.C, NSPLIT), 256, 0, st>>>(raw.p, part, HW, raw.N,
                                                                        (long)raw.C * HW);
    bn_stats_fin_kernel<<<(unsigned)cdiv_i(raw.C, 256), 256, 0, st>>>(part, stats, cnt, raw.C);
    long total = raw.numel();
    bn_apply_kernel<<<(unsigned)((total + 255) / 256), 256, 0, st>>>(
        raw.p, stats, p.g, p.bb, addsrc, raw.p, raw.C, HW, total);
    return raw;
}

static void up2_add(hipStream_t st, const Tensor& dst, const Tensor& src) {
    long total = dst.numel();
    up2_add_kernel<<<(unsigned)((total + 255) / 256), 256, 0, st>>>(dst.p, src.p, src.H, src.W, total);
}

static void run_head(hipStream_t st, Arena& S, float* stats, const Tensor& f,
                     const HeadP& h, float* outp) {
    Tensor t0 = cbr(st, S, S, stats, f, h.c0, 1);
    Tensor t1 = cbr(st, S, S, stats, t0, h.c1, 1);
    Tensor raw = run_conv(st, S, S, t1, h.pw, 255, 1, 1);
    long total = raw.numel();
    head_out_kernel<<<(unsigned)((total + 255) / 256), 256, 0, st>>>(raw.p, h.pb, outp,
                                                                     raw.H, raw.W, total);
}

extern "C" void kernel_launch(void* const* d_in, const int* in_sizes, int n_in,
                              void* d_out, int out_size, void* d_ws, size_t ws_size,
                              hipStream_t stream) {
    (void)in_sizes; (void)n_in; (void)out_size;

    // ---- parameter cursor (jax pytree: dicts flattened by sorted key) ----
    int ci = 1;  // d_in[0] = x
    auto nxt = [&]() { return (const float*)d_in[ci++]; };
    auto rd_cbr = [&](int cin, int cout, int k) {
        CbrP p; p.bb = nxt(); p.g = nxt(); p.w = nxt();  // keys: b, g, w
        p.cin = cin; p.cout = cout; p.k = k; return p;
    };
    auto rd_res = [&](int C) { ResP r; r.a = rd_cbr(C, C / 2, 1); r.b = rd_cbr(C / 2, C, 3); return r; };
    auto rd_head = [&]() {
        HeadP h; h.c0 = rd_cbr(128, 256, 3); h.c1 = rd_cbr(256, 256, 3);
        h.pb = nxt(); h.pw = nxt();  // keys: pb, pw
        return h;
    };

    // backbone (sorted keys: s0_0,s0_1,s1_0,s1_r0,s2_0,s2_r0,s2_r1,s3_0,s3_res,s4_0,s4_1,s4_r0)
    CbrP s0_0 = rd_cbr(3, 32, 3);
    CbrP s0_1 = rd_cbr(32, 64, 3);
    CbrP s1_0 = rd_cbr(64, 128, 3);
    ResP s1_r0 = rd_res(128);
    CbrP s2_0 = rd_cbr(128, 256, 3);
    ResP s2_r0 = rd_res(256);
    ResP s2_r1 = rd_res(256);
    CbrP s3_0 = rd_cbr(256, 512, 3);
    ResP s3_res0 = rd_res(512), s3_res1 = rd_res(512), s3_res2 = rd_res(512);
    CbrP s4_0 = rd_cbr(512, 1024, 3);
    CbrP s4_1 = rd_cbr(1024, 512, 1);
    ResP s4_r0 = rd_res(1024);
    // fpn (lat3,lat4,lat5,out3,out4,out5)
    CbrP lat3 = rd_cbr(256, 128, 1), lat4 = rd_cbr(512, 128, 1), lat5 = rd_cbr(512, 128, 1);
    CbrP out3 = rd_cbr(128, 128, 3), out4 = rd_cbr(128, 128, 3), out5 = rd_cbr(128, 128, 3);
    // heads (sorted: head_l, head_m, head_s)
    HeadP head_l = rd_head(), head_m = rd_head(), head_s = rd_head();

    // ---- workspace arenas ----
    char* base = (char*)d_ws;
    float* stats = (float*)base;                        // 2*1024 floats
    size_t off0 = 8192;
    size_t avail = ws_size > off0 ? ws_size - off0 : 0;
    size_t pcap = avail / 3;
    Arena P{base + off0, pcap, 0};                      // persistent (inter-stage)
    Arena S{base + off0 + pcap, avail - pcap, 0};       // scratch (reset per stage)

    Tensor X{(float*)d_in[0], 4, 3, 416, 416};

    // stage 1
    Tensor a0 = cbr(stream, S, S, stats, X, s0_0, 1);
    Tensor a1 = cbr(stream, S, P, stats, a0, s0_1, 2);
    S.reset();
    // stage 2
    Tensor b0 = cbr(stream, S, S, stats, a1, s1_0, 2);
    Tensor t  = cbr(stream, S, S, stats, b0, s1_r0.a, 1);
    Tensor h1 = cbr(stream, S, P, stats, t, s1_r0.b, 1, b0.p);
    S.reset();
    // stage 3 -> p3
    Tensor r = cbr(stream, S, S, stats, h1, s2_0, 2);
    { Tensor u = cbr(stream, S, S, stats, r, s2_r0.a, 1);
      r = cbr(stream, S, S, stats, u, s2_r0.b, 1, r.p); }
    Tensor p3{};
    { Tensor u = cbr(stream, S, S, stats, r, s2_r1.a, 1);
      p3 = cbr(stream, S, P, stats, u, s2_r1.b, 1, r.p); }
    S.reset();
    // stage 4 -> p4
    r = cbr(stream, S, S, stats, p3, s3_0, 2);
    { Tensor u = cbr(stream, S, S, stats, r, s3_res0.a, 1);
      r = cbr(stream, S, S, stats, u, s3_res0.b, 1, r.p); }
    { Tensor u = cbr(stream, S, S, stats, r, s3_res1.a, 1);
      r = cbr(stream, S, S, stats, u, s3_res1.b, 1, r.p); }
    Tensor p4{};
    { Tensor u = cbr(stream, S, S, stats, r, s3_res2.a, 1);
      p4 = cbr(stream, S, P, stats, u, s3_res2.b, 1, r.p); }
    S.reset();
    // stage 5 -> p5
    Tensor e0 = cbr(stream, S, S, stats, p4, s4_0, 2);
    { Tensor u = cbr(stream, S, S, stats, e0, s4_r0.a, 1);
      r = cbr(stream, S, S, stats, u, s4_r0.b, 1, e0.p); }
    Tensor p5 = cbr(stream, S, P, stats, r, s4_1, 1);
    S.reset();
    // FPN
    Tensor n5 = cbr(stream, S, P, stats, p5, lat5, 1);
    Tensor n4 = cbr(stream, S, P, stats, p4, lat4, 1);
    up2_add(stream, n4, n5);
    Tensor n3 = cbr(stream, S, P, stats, p3, lat3, 1);
    up2_add(stream, n3, n4);
    n3 = cbr(stream, S, P, stats, n3, out3, 1);
    n4 = cbr(stream, S, P, stats, n4, out4, 1);
    n5 = cbr(stream, S, P, stats, n5, out5, 1);
    S.reset();
    // heads: output tuple order (head_s, head_m, head_l)
    float* out = (float*)d_out;
    const long off_s = 0;
    const long off_m = (long)4 * 3 * 52 * 52 * 85;            // 2,758,080
    const long off_l = off_m + (long)4 * 3 * 26 * 26 * 85;    // +689,520
    run_head(stream, S, stats, n3, head_s, out + off_s); S.reset();
    run_head(stream, S, stats, n4, head_m, out + off_m); S.reset();
    run_head(stream, S, stats, n5, head_l, out + off_l); S.reset();
}
